// SampleModel_86921548137051
// MI455X (gfx1250) — matrix operand order repaired and checked by reading, not compiled, not run on hardware
//
#include <hip/hip_runtime.h>

typedef __attribute__((ext_vector_type(16))) __bf16 v16bf;
typedef __attribute__((ext_vector_type(8)))  __bf16 v8bf;
typedef __attribute__((ext_vector_type(8)))  float  v8f;

static constexpr int Bn = 32768;   // batch rows
static constexpr int Kc = 8192;    // centroids
static constexpr int Dd = 384;     // feature dim  (12 k-steps of 32)
static constexpr float TEMP_INV = 1.0f / 0.07f;

// ---------------------------------------------------------------------------
// Zero d_out and cent_sum accumulator (graph-replay safe: runs every launch)
// ---------------------------------------------------------------------------
__global__ void zero_accum(float* __restrict__ out, float* __restrict__ csum) {
    int i = blockIdx.x * blockDim.x + threadIdx.x;
    if (i < Kc) csum[i] = 0.0f;
    if (i == 0) out[0] = 0.0f;
}

// ---------------------------------------------------------------------------
// features f32 -> bf16 (elementwise)
// ---------------------------------------------------------------------------
__global__ void f32_to_bf16(const float* __restrict__ in, __bf16* __restrict__ out, int n) {
    int i = blockIdx.x * blockDim.x + threadIdx.x;
    if (i < n) out[i] = (__bf16)in[i];
}

// ---------------------------------------------------------------------------
// L2-normalize centroid rows, emit bf16. One wave32 per row (D=384 = 12*32).
// ---------------------------------------------------------------------------
__global__ void __launch_bounds__(256) normalize_centroids(
        const float* __restrict__ cent, __bf16* __restrict__ out) {
    const int wave = threadIdx.x >> 5;
    const int lane = threadIdx.x & 31;
    const int row  = blockIdx.x * 8 + wave;
    const float* r = cent + (size_t)row * Dd;
    float v[12];
    float ss = 0.0f;
#pragma unroll
    for (int i = 0; i < 12; ++i) { v[i] = r[lane + i * 32]; ss += v[i] * v[i]; }
#pragma unroll
    for (int off = 16; off > 0; off >>= 1) ss += __shfl_xor(ss, off, 32);
    const float scale = 1.0f / fmaxf(sqrtf(ss), 1e-12f);
    __bf16* o = out + (size_t)row * Dd;
#pragma unroll
    for (int i = 0; i < 12; ++i) o[lane + i * 32] = (__bf16)(v[i] * scale);
}

// ---------------------------------------------------------------------------
// A fragment (16x32 bf16): lane = row (lane&15), two 16B chunks:
//   K in [kbase, kbase+8) and [kbase+8, kbase+16), kbase = kd + (lane>>4)*16
// ---------------------------------------------------------------------------
__device__ __forceinline__ v16bf load_a_frag(const __bf16* p) {
    v8bf lo = *(const v8bf*)(p);
    v8bf hi = *(const v8bf*)(p + 8);
    v16bf a;
#pragma unroll
    for (int e = 0; e < 8; ++e) { a[e] = lo[e]; a[e + 8] = hi[e]; }
    return a;
}

// ---------------------------------------------------------------------------
// GEMM1: features @ c^T with fused row-max / arg-max (logit domain).
// Wave owns a 16-row strip; 12 A fragments (96 VGPRs) pinned in registers.
// B is explicitly double-buffered (bcur/bnxt, 64 VGPRs) so 8 loads for
// k-step ks+1 are in flight while the 4 WMMAs of k-step ks execute.
// ---------------------------------------------------------------------------
__global__ void __launch_bounds__(256) gemm_rowmax(
        const __bf16* __restrict__ A,   // [Bn, Dd]
        const __bf16* __restrict__ Cb,  // [Kc, Dd] (normalized, bf16)
        float* __restrict__ rmax_out,   // [Bn]
        int*   __restrict__ rarg_out) { // [Bn]
    const int wave = threadIdx.x >> 5;
    const int lane = threadIdx.x & 31;
    const int ln = lane & 15, lh = lane >> 4;
    const int r0 = blockIdx.x * 128 + wave * 16;

    // hoist the full A strip for this wave: 12 fragments, loop-invariant
    const __bf16* arow = A + (size_t)(r0 + ln) * Dd + lh * 16;
    v16bf afrag[12];
#pragma unroll
    for (int ks = 0; ks < 12; ++ks) afrag[ks] = load_a_frag(arow + ks * 32);

    float rmax[8]; int rarg[8];
#pragma unroll
    for (int v = 0; v < 8; ++v) { rmax[v] = -3.4e38f; rarg[v] = 0; }

    // per-lane B base for column (col + ln): Cb[(col+ln)*Dd + kd + lh*16]
    const __bf16* bbase = Cb + (size_t)ln * Dd + lh * 16;

    for (int ct = 0; ct < Kc / 64; ++ct) {
        const int col0 = ct * 64;
        if (ct + 1 < Kc / 64)  // speculative prefetch of next B supertile
            __builtin_prefetch(bbase + (size_t)(col0 + 64) * Dd, 0, 1);

        v8f acc[4];
#pragma unroll
        for (int s = 0; s < 4; ++s)
#pragma unroll
            for (int v = 0; v < 8; ++v) acc[s][v] = 0.0f;

        v16bf bcur[4], bnxt[4];
#pragma unroll
        for (int s = 0; s < 4; ++s)
            bcur[s] = *(const v16bf*)(bbase + (size_t)(col0 + s * 16) * Dd);

#pragma unroll
        for (int ks = 0; ks < 12; ++ks) {
            if (ks + 1 < 12) {   // issue next k-step's 8 B loads first
#pragma unroll
                for (int s = 0; s < 4; ++s)
                    bnxt[s] = *(const v16bf*)(bbase + (size_t)(col0 + s * 16) * Dd
                                              + (ks + 1) * 32);
            }
#pragma unroll
            for (int s = 0; s < 4; ++s)
                acc[s] = __builtin_amdgcn_wmma_f32_16x16x32_bf16(
                             false, afrag[ks], false, bcur[s], (short)0,
                             acc[s], false, false);
#pragma unroll
            for (int s = 0; s < 4; ++s) bcur[s] = bnxt[s];
        }

        // running per-lane max/argmax over this lane's columns
#pragma unroll
        for (int s = 0; s < 4; ++s) {
            const int cidx = col0 + s * 16 + ln;
#pragma unroll
            for (int v = 0; v < 8; ++v) {
                float x = acc[s][v];
                if (x > rmax[v]) { rmax[v] = x; rarg[v] = cidx; }
            }
        }
    }

    // cross-lane argmax reduce within each 16-lane half (C layout: masks <= 8)
#pragma unroll
    for (int v = 0; v < 8; ++v) {
        float mv = rmax[v]; int av = rarg[v];
#pragma unroll
        for (int off = 8; off >= 1; off >>= 1) {
            float om = __shfl_xor(mv, off, 32);
            int   oa = __shfl_xor(av, off, 32);
            if (om > mv || (om == mv && oa < av)) { mv = om; av = oa; }
        }
        if (ln == 0) {
            const int row = r0 + lh * 8 + v;  // lanes 0 & 16 cover rows v and v+8
            rmax_out[row] = mv;
            rarg_out[row] = av;
        }
    }
}

// ---------------------------------------------------------------------------
// GEMM2: c @ c^T with fused exp(./T) + column-sum. Dual of GEMM1: wave owns
// 16 columns, holds their 12 B fragments (96 VGPRs) in registers for the
// whole 1024-row slice; streaming A is explicitly double-buffered.
// ---------------------------------------------------------------------------
__global__ void __launch_bounds__(256) gemm_centsum(
        const __bf16* __restrict__ Cb,     // [Kc, Dd]
        float* __restrict__ csum_out) {    // [Kc], pre-zeroed
    const int wave = threadIdx.x >> 5;
    const int lane = threadIdx.x & 31;
    const int ln = lane & 15, lh = lane >> 4;
    const int c0 = blockIdx.x * 128 + wave * 16;   // 16 columns per wave
    const int rbeg = blockIdx.y * (Kc / 8);        // 1024-row slice -> 64 tiles

    // hoist B fragments for this wave's columns: loop-invariant over rows
    const __bf16* brow = Cb + (size_t)(c0 + ln) * Dd + lh * 16;
    v16bf bfrag[12];
#pragma unroll
    for (int ks = 0; ks < 12; ++ks) bfrag[ks] = *(const v16bf*)(brow + ks * 32);

    float csum = 0.0f;

    for (int rt = 0; rt < (Kc / 8) / 16; ++rt) {
        const int r0 = rbeg + rt * 16;
        const __bf16* arow = Cb + (size_t)(r0 + ln) * Dd + lh * 16;

        v8f acc;
#pragma unroll
        for (int v = 0; v < 8; ++v) acc[v] = 0.0f;

        v16bf acur = load_a_frag(arow);
        v16bf anxt;
#pragma unroll
        for (int ks = 0; ks < 12; ++ks) {
            if (ks + 1 < 12) anxt = load_a_frag(arow + (ks + 1) * 32);
            acc = __builtin_amdgcn_wmma_f32_16x16x32_bf16(
                      false, acur, false, bfrag[ks], (short)0, acc, false, false);
            acur = anxt;
        }
#pragma unroll
        for (int v = 0; v < 8; ++v)
            csum += __expf(acc[v] * TEMP_INV);
    }

    // halves hold disjoint row sets of the same column: fold lane <-> lane+16
    csum += __shfl_xor(csum, 16, 32);
    if (lh == 0) atomicAdd(&csum_out[c0 + ln], csum);
}

// ---------------------------------------------------------------------------
// Final scalar: J_b = m - log(exp(m) + cent_sum[arg]),  out = -mean(J)
// ---------------------------------------------------------------------------
__global__ void __launch_bounds__(256) final_loss(
        const float* __restrict__ rmax, const int* __restrict__ rarg,
        const float* __restrict__ csum, float* __restrict__ out) {
    __shared__ float sdata[256];
    const int b = blockIdx.x * blockDim.x + threadIdx.x;
    const float m  = rmax[b] * TEMP_INV;
    const float cs = csum[rarg[b]];
    const float J  = m - logf(__expf(m) + cs);   // BALANCE = 1
    sdata[threadIdx.x] = -J * (1.0f / (float)Bn);
    __syncthreads();
#pragma unroll
    for (int off = 128; off > 0; off >>= 1) {
        if (threadIdx.x < off) sdata[threadIdx.x] += sdata[threadIdx.x + off];
        __syncthreads();
    }
    if (threadIdx.x == 0) atomicAdd(out, sdata[0]);
}

// ---------------------------------------------------------------------------
extern "C" void kernel_launch(void* const* d_in, const int* in_sizes, int n_in,
                              void* d_out, int out_size, void* d_ws, size_t ws_size,
                              hipStream_t stream) {
    const float* features  = (const float*)d_in[0];  // [Bn, Dd]
    const float* centroids = (const float*)d_in[1];  // [Kc, Dd]
    float* out = (float*)d_out;

    char* ws = (char*)d_ws;
    __bf16* feat_bf = (__bf16*)ws;                                   // 25 MB
    __bf16* cent_bf = (__bf16*)(ws + (size_t)Bn * Dd * 2);           //  6.3 MB
    float*  rmax    = (float*)(ws + (size_t)Bn * Dd * 2 + (size_t)Kc * Dd * 2);
    int*    rarg    = (int*)((char*)rmax + (size_t)Bn * 4);
    float*  csum    = (float*)((char*)rarg + (size_t)Bn * 4);

    zero_accum<<<(Kc + 255) / 256, 256, 0, stream>>>(out, csum);
    f32_to_bf16<<<(Bn * Dd + 255) / 256, 256, 0, stream>>>(features, feat_bf, Bn * Dd);
    normalize_centroids<<<Kc / 8, 256, 0, stream>>>(centroids, cent_bf);

    gemm_rowmax<<<Bn / 128, 256, 0, stream>>>(feat_bf, cent_bf, rmax, rarg);

    dim3 g2(Kc / 128, 8);  // 64 column groups x 8 row slices = 512 blocks
    gemm_centsum<<<g2, 256, 0, stream>>>(cent_bf, csum);

    final_loss<<<Bn / 256, 256, 0, stream>>>(rmax, rarg, csum, out);
}
